// InfoNCELoss_80599356277327
// MI455X (gfx1250) — matrix operand order repaired and checked by reading, not compiled, not run on hardware
//
#include <hip/hip_runtime.h>
#include <hip/hip_bf16.h>

// InfoNCE loss, N=16384, D=64, E=262144, T=0.1
// ws layout: [ zn bf16 N*64 | num f32 N | den f32 N | bitmask N*N/8 ]

typedef __attribute__((ext_vector_type(16))) __bf16 v16bf;
typedef __attribute__((ext_vector_type(8)))  __bf16 v8bf;
typedef __attribute__((ext_vector_type(8)))  float  v8f;
typedef __attribute__((ext_vector_type(4)))  unsigned int u32x4;
typedef __attribute__((ext_vector_type(8)))  int i32x8;
typedef __attribute__((ext_vector_type(4)))  int i32x4;

#define DFEAT 64
#define TEMP_INV 10.0f
// exp(TEMP_INV * x) == exp2(EXP_SCALE * x): one v_pk_mul instead of two per pair
#define EXP_SCALE 14.426950408889634f
#define CHUNK 128                // columns staged per TDM transfer (16 KB)
#define ROWB 144                 // 128B row + 16B TDM pad -> conflict-free LDS banks
#define BUFB (CHUNK * ROWB)      // 18432 B per buffer (x2 double buffer = 36 KB)

#ifndef __has_builtin
#define __has_builtin(x) 0
#endif
#if __has_builtin(__builtin_amdgcn_tensor_load_to_lds) && \
    __has_builtin(__builtin_amdgcn_s_wait_tensorcnt)
#define USE_TDM 1
#else
#define USE_TDM 0
#endif

// Raw v_exp_f32: argument range here is [-14.43, +14.43], far from the -126
// flush threshold, so libm exp2f's denormal fixup (v_cmp + cndmask per
// element) is dead weight on the 268M-element hot path.
#if __has_builtin(__builtin_amdgcn_exp2f)
#define FAST_EXP2(x) __builtin_amdgcn_exp2f(x)
#else
#define FAST_EXP2(x) exp2f(x)
#endif

// ---------- Kernel 1: row-normalize fp32 -> bf16 (one wave32 per row) ----------
__global__ void k_normalize(const float* __restrict__ z, __bf16* __restrict__ zn, int n) {
  int row  = (blockIdx.x * blockDim.x + threadIdx.x) >> 5;
  int lane = threadIdx.x & 31;
  if (row >= n) return;
  float2 v = ((const float2*)(z + (size_t)row * DFEAT))[lane];
  float ss = v.x * v.x + v.y * v.y;
  #pragma unroll
  for (int m = 16; m >= 1; m >>= 1) ss += __shfl_xor(ss, m, 32);
  float rn = rsqrtf(ss);
  __bf16* o = zn + (size_t)row * DFEAT + lane * 2;
  o[0] = (__bf16)(v.x * rn);
  o[1] = (__bf16)(v.y * rn);
}

// ---------- Kernel 2: init numerator with diagonal term, mark diagonal bits ----------
__global__ void k_init(float* __restrict__ num, unsigned int* __restrict__ mask, int n) {
  int i = blockIdx.x * blockDim.x + threadIdx.x;
  if (i >= n) return;
  num[i] = FAST_EXP2(EXP_SCALE);  // cos(z_i,z_i)=1 exactly after normalize
  unsigned long long bit = (unsigned long long)i * (unsigned long long)n + (unsigned long long)i;
  atomicOr(&mask[bit >> 5], 1u << (unsigned)(bit & 31));
}

#if USE_TDM
// Issue one TDM transfer: CHUNK*DFEAT bf16 (16 KB contiguous) -> LDS with
// 16B padding every 128B row (D# pad_interval=32 DWORDs, pad_amount=4 DWORDs).
__device__ __forceinline__ void tdm_load_chunk(const __bf16* gsrc, unsigned lds_off) {
  unsigned long long ga = (unsigned long long)(uintptr_t)gsrc;
  const int elems = CHUNK * DFEAT;  // 8192 elements, data_size = 2B
  u32x4 g0;
  g0.x = 1u;                                     // count=1 (valid), user mode
  g0.y = lds_off;                                // lds_addr (bytes)
  g0.z = (unsigned)ga;                           // global_addr[31:0]
  g0.w = ((unsigned)(ga >> 32) & 0x01FFFFFFu)    // global_addr[56:32]
         | (2u << 30);                           // type = 2
  i32x8 g1;
  g1[0] = (1 << 16)        // data_size: 2 bytes
        | (1 << 20)        // pad_enable
        | (4 << 22)        // pad_interval: every 32 DWORDs (=128B = one row)
        | (3 << 25);       // pad_amount: 4 DWORDs (=16B)
  g1[1] = (elems & 0xFFFF) << 16;                 // tensor_dim0[15:0]
  g1[2] = ((elems >> 16) & 0xFFFF) | (1 << 16);   // tensor_dim0[31:16] | tensor_dim1=1
  g1[3] = (elems & 0xFFFF) << 16;                 // tile_dim0 = 8192
  g1[4] = 1;                                      // tile_dim1 = 1
  g1[5] = elems;                                  // tensor_dim0_stride[31:0]
  g1[6] = 0;
  g1[7] = 0;
  i32x4 z4 = {0, 0, 0, 0};
#if __clang_major__ >= 23
  i32x8 z8 = {0, 0, 0, 0, 0, 0, 0, 0};
  __builtin_amdgcn_tensor_load_to_lds(g0, g1, z4, z4, z8, 0);
#else
  __builtin_amdgcn_tensor_load_to_lds(g0, g1, z4, z4, 0);
#endif
}
#endif

// ---------- Kernel 3: denominator = rowsum(exp(10 * Zn Zn^T)) ----------
// 128 threads = 4 waves; each wave owns 16 rows (A resident in VGPRs).
// B columns staged 128 at a time into LDS by the Tensor Data Mover (wave 0),
// double-buffered; all 4 waves consume fragments via ds_load.
__global__ __launch_bounds__(128)
void k_denom(const __bf16* __restrict__ zn, float* __restrict__ den, int n) {
  __shared__ alignas(32) unsigned char smem[2][BUFB];
  const int lane = threadIdx.x & 31;
  const int wave = threadIdx.x >> 5;
  const int m0   = blockIdx.x * 64 + wave * 16;
  const int half = lane >> 4;      // 0: lanes 0-15, 1: lanes 16-31
  const int l15  = lane & 15;

  // A fragments (ISA 16-bit A 16x32 layout): lane = row m0+l15,
  // K chunks {base..base+7, base+16..base+23}, base = 8*half.
  const __bf16* arow = zn + (size_t)(m0 + l15) * DFEAT;
  const int abase = half * 8;
  union { v16bf v; v8bf h[2]; } a0, a1;
  a0.h[0] = *(const v8bf*)(arow + abase);
  a0.h[1] = *(const v8bf*)(arow + abase + 16);
  a1.h[0] = *(const v8bf*)(arow + abase + 32);
  a1.h[1] = *(const v8bf*)(arow + abase + 48);

  float sums[8];
  #pragma unroll
  for (int r = 0; r < 8; ++r) sums[r] = 0.f;

  const int nC = n / CHUNK;
  const unsigned bk_byte = (unsigned)half * 16;  // lane's K-start byte within a row

#if USE_TDM
  if (wave == 0) tdm_load_chunk(zn, (unsigned)(uintptr_t)&smem[0][0]);
#endif

  for (int c = 0; c < nC; ++c) {
    const int bi = c & 1;
#if USE_TDM
    if (wave == 0) {
      if (c + 1 < nC) {
        tdm_load_chunk(zn + (size_t)(c + 1) * CHUNK * DFEAT,
                       (unsigned)(uintptr_t)&smem[bi ^ 1][0]);
        __builtin_amdgcn_s_wait_tensorcnt(1);   // chunk c complete (in-order)
      } else {
        __builtin_amdgcn_s_wait_tensorcnt(0);
      }
    }
    __syncthreads();
#else
    {   // cooperative staging fallback, same padded pitch
      const __bf16* src = zn + (size_t)c * CHUNK * DFEAT;
      for (int i = threadIdx.x; i < CHUNK * 8; i += 128) {
        int r = i >> 3, seg = i & 7;
        __builtin_prefetch(src + (size_t)CHUNK * DFEAT + r * DFEAT + seg * 8, 0, 3);
        *(v8bf*)(&smem[bi][r * ROWB + seg * 16]) =
            *(const v8bf*)(src + (size_t)r * DFEAT + seg * 8);
      }
      __syncthreads();
    }
#endif
    #pragma unroll
    for (int t = 0; t < CHUNK / 16; ++t) {
      // B 32x16 layout: lane = column (chunk-row t*16+l15), K chunks {8*half..+7, 16+8*half..+7}
      const unsigned char* brow = &smem[bi][(unsigned)(t * 16 + l15) * ROWB];
      union { v16bf v; v8bf h[2]; } b0, b1;
      b0.h[0] = *(const v8bf*)(brow + bk_byte);
      b0.h[1] = *(const v8bf*)(brow + bk_byte + 32);
      b1.h[0] = *(const v8bf*)(brow + 64 + bk_byte);
      b1.h[1] = *(const v8bf*)(brow + 64 + bk_byte + 32);
      v8f acc = {};
      acc = __builtin_amdgcn_wmma_f32_16x16x32_bf16(false, a0.v, false, b0.v, (short)0, acc, false, false);
      acc = __builtin_amdgcn_wmma_f32_16x16x32_bf16(false, a1.v, false, b1.v, (short)0, acc, false, false);
      // acc[r] = sim[m0 + 8*half + r][c*CHUNK + t*16 + l15]
      #pragma unroll
      for (int r = 0; r < 8; ++r) sums[r] += FAST_EXP2(EXP_SCALE * acc[r]);
    }
    __syncthreads();   // all waves done with buf bi before TDM overwrites it
  }

  // reduce over the 16 lanes of each half (masks 1,2,4,8 stay within half)
  #pragma unroll
  for (int r = 0; r < 8; ++r) {
    float s = sums[r];
    s += __shfl_xor(s, 1, 32);
    s += __shfl_xor(s, 2, 32);
    s += __shfl_xor(s, 4, 32);
    s += __shfl_xor(s, 8, 32);
    sums[r] = s;
  }
  if (l15 == 0) {
    #pragma unroll
    for (int r = 0; r < 8; ++r) den[m0 + half * 8 + r] = sums[r];
  }
}

// ---------- Kernel 4: sparse numerator over edges, dedup via atomicOr bitmask ----------
__global__ void k_edges(const int* __restrict__ ei, const __bf16* __restrict__ zn,
                        float* __restrict__ num, unsigned int* __restrict__ mask,
                        int n, int e) {
  int t = blockIdx.x * blockDim.x + threadIdx.x;
  if (t >= e) return;
  int a = ei[t];
  int b = ei[e + t];
  unsigned long long bit = (unsigned long long)a * (unsigned long long)n + (unsigned long long)b;
  unsigned int bm  = 1u << (unsigned)(bit & 31);
  unsigned int old = atomicOr(&mask[bit >> 5], bm);
  if (old & bm) return;   // duplicate edge or diagonal self-loop: already counted
  const __bf16* ra = zn + (size_t)a * DFEAT;
  const __bf16* rb = zn + (size_t)b * DFEAT;
  float dot = 0.f;
  #pragma unroll
  for (int k = 0; k < DFEAT; ++k) dot += (float)ra[k] * (float)rb[k];
  atomicAdd(&num[a], FAST_EXP2(EXP_SCALE * dot));
}

// ---------- Kernel 5: loss = -mean(log(num/den)) ----------
__global__ void k_loss(const float* __restrict__ num, const float* __restrict__ den,
                       float* __restrict__ out, int n) {
  __shared__ float sm[256];
  float s = 0.f;
  for (int i = threadIdx.x; i < n; i += 256)
    s += __logf(num[i] / den[i]);
  sm[threadIdx.x] = s;
  __syncthreads();
  for (int w = 128; w >= 1; w >>= 1) {
    if (threadIdx.x < w) sm[threadIdx.x] += sm[threadIdx.x + w];
    __syncthreads();
  }
  if (threadIdx.x == 0) out[0] = -sm[0] / (float)n;
}

extern "C" void kernel_launch(void* const* d_in, const int* in_sizes, int n_in,
                              void* d_out, int out_size, void* d_ws, size_t ws_size,
                              hipStream_t stream) {
  const float* z  = (const float*)d_in[0];
  const int*   ei = (const int*)d_in[1];
  float*       out = (float*)d_out;
  const int n = in_sizes[0] / DFEAT;   // 16384
  const int e = in_sizes[1] / 2;       // 262144

  char* ws = (char*)d_ws;
  __bf16* zn = (__bf16*)ws;
  size_t off = ((size_t)n * DFEAT * sizeof(__bf16) + 255) & ~(size_t)255;
  float* num = (float*)(ws + off); off += (size_t)n * sizeof(float);
  float* den = (float*)(ws + off); off += (size_t)n * sizeof(float);
  off = (off + 255) & ~(size_t)255;
  unsigned int* mask = (unsigned int*)(ws + off);
  size_t maskBytes = ((((size_t)n * (size_t)n + 7) / 8) + 3) & ~(size_t)3;  // 32 MB

  hipMemsetAsync(mask, 0, maskBytes, stream);

  k_normalize<<<(n + 7) / 8, 256, 0, stream>>>(z, zn, n);
  k_init<<<(n + 255) / 256, 256, 0, stream>>>(num, mask, n);
  k_denom<<<n / 64, 128, 0, stream>>>(zn, den, n);
  k_edges<<<(e + 255) / 256, 256, 0, stream>>>(ei, zn, num, mask, n, e);
  k_loss<<<1, 256, 0, stream>>>(num, den, out, n);
}
